// C3WithGLCM_67843303408164
// MI455X (gfx1250) — matrix operand order repaired and checked by reading, not compiled, not run on hardware
//
#include <hip/hip_runtime.h>

typedef __attribute__((ext_vector_type(16))) _Float16 v16h;
typedef __attribute__((ext_vector_type(8)))  _Float16 v8h;
typedef __attribute__((ext_vector_type(8)))  float    v8f;

#define NCOL 8192   // N = B*H*W (activation matrix columns)
#define HWSZ 4096   // H*W
#define NTIL 512    // NCOL/16 n-tiles

// -------------------------------------------------------------------------
// Fragment-native blocked layout for activation (B) matrices:
//   element (k, n) -> tile (kt=k/32, nt=n/16), lane = (k%32 & 16) + n%16,
//   half j = k%32 & 15.  Each lane's 16 halves are 32 contiguous bytes, so a
//   B fragment is exactly two coalesced global_load_b128 per lane.
// Same contiguous k order as the packed A weights (lanes 0-15: K 0-15,
// lanes 16-31: K 16-31, one column per lane).
// -------------------------------------------------------------------------
__device__ __forceinline__ size_t bidx(int k, int n) {
  int kt = k >> 5, kr = k & 31;
  int nt = n >> 4, nc = n & 15;
  return ((size_t)(kt * NTIL + nt) * 32 + (kr & 16) + nc) * 16 + (kr & 15);
}

// -------------------------------------------------------------------------
// Weight pack: fp32 [M][K] -> fp16 fragment-native A layout (zero pad in K).
// Contiguous k order, same as B: lane l holds row l%16; lanes 0-15 carry
// K 0-15, lanes 16-31 carry K 16-31:  K(j) = 32*kt + 16*(l/16) + j.
// -------------------------------------------------------------------------
__global__ void pack_weights_kernel(const float* __restrict__ src,
                                    _Float16* __restrict__ dst,
                                    int M, int K, int Kp) {
  int idx = blockIdx.x * 256 + threadIdx.x;
  if (idx >= M * Kp) return;
  int m = idx / Kp, kp = idx - m * Kp;
  float v = (kp < K) ? src[m * K + kp] : 0.f;
  int KT = Kp >> 5;
  int mt = m >> 4, mr = m & 15;
  int kt = kp >> 5, kk = kp & 31;
  int lane = (kk & 16) + mr;
  int j    = kk & 15;
  dst[((size_t)(mt * KT + kt) * 32 + lane) * 16 + j] = (_Float16)v;
}

// -------------------------------------------------------------------------
// GLCM features + build fp16 activation matrix xch (64 K-rows, blocked):
//   rows 0..31 = x, rows 32..35 = GLCM feats, rows 36..63 = zero pad.
// One wave = one pixel (lane = channel); 20 directed pairs of the 3x3 patch;
// multiplicity trick replaces the 256-bin histogram.
// -------------------------------------------------------------------------
__global__ void glcm_pack_kernel(const float* __restrict__ x,
                                 _Float16* __restrict__ xch) {
  const int lane = threadIdx.x & 31;     // channel
  const int tp   = threadIdx.x >> 5;     // wave id = pixel in block
  const int n    = blockIdx.x * 8 + tp;  // column (= b*4096 + hw)
  const int b  = n >> 12;
  const int hw = n & 4095;
  const int h  = hw >> 6, w = hw & 63;
  const float* xc = x + ((size_t)b * 32 + lane) * HWSZ;

  float patch[9];
#pragma unroll
  for (int r = 0; r < 3; ++r)
#pragma unroll
    for (int c = 0; c < 3; ++c) {
      int hh = min(max(h + r - 1, 0), 63);
      int ww = min(max(w + c - 1, 0), 63);
      patch[r * 3 + c] = xc[hh * 64 + ww];
    }

  int q[9];
#pragma unroll
  for (int i = 0; i < 9; ++i) {
    int v = (int)(patch[i] * 15.0f);
    q[i] = min(max(v, 0), 15);
  }

  static constexpr int I1[20] = {0,1,3,4,6,7, 0,1,2,3,4,5, 0,1,3,4, 1,2,4,5};
  static constexpr int I2[20] = {1,2,4,5,7,8, 3,4,5,6,7,8, 4,5,7,8, 3,4,6,7};

  int pv[20];
  float contrast = 0.f, homog = 0.f;
#pragma unroll
  for (int t = 0; t < 20; ++t) {
    int a = q[I1[t]], bq = q[I2[t]];
    pv[t] = a * 16 + bq;
    int d = a - bq;
    int ad = d < 0 ? -d : d;
    contrast += (float)(d * d);
    homog += __builtin_amdgcn_rcpf((float)(1 + ad));   // v_rcp_f32, ~1ulp
  }
  contrast *= 0.05f;
  homog    *= 0.05f;

  float esum = 0.f, entropy = 0.f;
#pragma unroll
  for (int k = 0; k < 20; ++k) {
    int m = 0;
#pragma unroll
    for (int j = 0; j < 20; ++j) m += (pv[j] == pv[k]) ? 1 : 0;
    esum += (float)m;
    entropy -= 0.05f * __logf((float)m * 0.05f + 1e-6f);
  }
  float energy = esum * (1.0f / 400.0f);

  float f[4] = {contrast, energy, entropy, homog};
#pragma unroll
  for (int i = 0; i < 4; ++i) {
    float v = f[i];
#pragma unroll
    for (int off = 16; off >= 1; off >>= 1) v += __shfl_xor(v, off, 32);
    f[i] = v * (1.0f / 32.0f);
  }

  xch[bidx(lane, n)] = (_Float16)patch[4];
  if (lane < 4)  xch[bidx(32 + lane, n)] = (_Float16)f[lane];
  if (lane < 28) xch[bidx(36 + lane, n)] = (_Float16)0.f;
}

// -------------------------------------------------------------------------
// im2col for the 3x3 conv (zero pad), writing the blocked layout.
// Row order k = ic*9 + tap matches OIHW weight flattening; rows 144..159 pad.
// -------------------------------------------------------------------------
__global__ void im2col3x3_kernel(const _Float16* __restrict__ hsrc,
                                 _Float16* __restrict__ hcol) {
  int idx = blockIdx.x * 256 + threadIdx.x;   // 160*8192 elements
  int k = idx / NCOL, n = idx - k * NCOL;
  _Float16 v = (_Float16)0.f;
  if (k < 144) {
    int ic = k / 9, t = k - ic * 9;
    int dy = t / 3 - 1, dx = t % 3 - 1;
    int hw = n & 4095;
    int h = hw >> 6, w = hw & 63;
    int y = h + dy, xx = w + dx;
    if ((unsigned)y < 64u && (unsigned)xx < 64u)
      v = hsrc[bidx(ic, (n - hw) + y * 64 + xx)];
  }
  hcol[bidx(k, n)] = v;
}

// -------------------------------------------------------------------------
// Wave32 WMMA GEMM, fully unrolled (MT = M/16, KT = Kp/32 template params).
// Each wave owns one 16-wide N-tile and accumulates ALL M-tiles for it, so
// the activation (B) matrix is read exactly once: 2 coalesced b128 per k-step.
// Epilogue mode (residual / fp16-blocked out / fp32-NCHW out) is compile-time
// so no per-element pointer tests or branches survive.
// -------------------------------------------------------------------------
template<int MT, int KT, bool HAS_RES, bool HAS_OUTH, bool HAS_OUTF>
__global__ void gemm_wmma_kernel(const _Float16* __restrict__ A,
                                 const _Float16* __restrict__ Bm,
                                 const float* __restrict__ scale,
                                 const float* __restrict__ bias,
                                 const _Float16* __restrict__ resid,
                                 _Float16* __restrict__ outH, int ocOff,
                                 float* __restrict__ outF) {
  const int lane = threadIdx.x & 31;
  const int wv   = threadIdx.x >> 5;
  const int nt   = blockIdx.x * 8 + wv;       // 0..511
  const int col  = lane & 15;
  const int half = lane >> 4;

  v8f acc[MT];
#pragma unroll
  for (int m = 0; m < MT; ++m) acc[m] = (v8f){};

#pragma unroll
  for (int kt = 0; kt < KT; ++kt) {
    union { v16h v; v8h h[2]; } bf;
    const _Float16* bp = Bm + ((size_t)(kt * NTIL + nt) * 32 + lane) * 16;
    bf.h[0] = *(const v8h*)(bp);
    bf.h[1] = *(const v8h*)(bp + 8);
#pragma unroll
    for (int m = 0; m < MT; ++m) {
      union { v16h v; v8h h[2]; } af;
      const _Float16* ap = A + ((size_t)(m * KT + kt) * 32 + lane) * 16;
      af.h[0] = *(const v8h*)(ap);
      af.h[1] = *(const v8h*)(ap + 8);
      acc[m] = __builtin_amdgcn_wmma_f32_16x16x32_f16(false, af.v, false, bf.v,
                                                      (short)0, acc[m], false, false);
    }
  }

  // C/D layout: VGPR r, lanes 0-15 -> row r, lanes 16-31 -> row r+8
  const int mloc = half * 8;
  const int n    = nt * 16 + col;
#pragma unroll
  for (int m = 0; m < MT; ++m) {
#pragma unroll
    for (int r = 0; r < 8; ++r) {
      int oc = m * 16 + mloc + r;
      float v = acc[m][r] * scale[oc] + bias[oc];
      // SiLU via fast reciprocal: exp + add + v_rcp_f32 + mul
      v = v * __builtin_amdgcn_rcpf(1.0f + __expf(-v));
      if constexpr (HAS_RES)  v += (float)resid[bidx(oc, n)];
      if constexpr (HAS_OUTH) outH[bidx(oc + ocOff, n)] = (_Float16)v;
      if constexpr (HAS_OUTF) {                          // NCHW fp32 final
        int bb = n >> 12, hw = n & 4095;
        outF[((size_t)bb * (MT * 16) + oc) * HWSZ + hw] = v;
      }
    }
  }
}

// -------------------------------------------------------------------------
extern "C" void kernel_launch(void* const* d_in, const int* in_sizes, int n_in,
                              void* d_out, int out_size, void* d_ws, size_t ws_size,
                              hipStream_t stream) {
  (void)in_sizes; (void)n_in; (void)out_size; (void)ws_size;
  const float* x    = (const float*)d_in[0];
  const float* Wcv1 = (const float*)d_in[1];
  const float* scv1 = (const float*)d_in[2];
  const float* bcv1 = (const float*)d_in[3];
  const float* Wcv2 = (const float*)d_in[4];
  const float* scv2 = (const float*)d_in[5];
  const float* bcv2 = (const float*)d_in[6];
  const float* Wcv3 = (const float*)d_in[7];
  const float* scv3 = (const float*)d_in[8];
  const float* bcv3 = (const float*)d_in[9];
  const float* Wm1  = (const float*)d_in[10];
  const float* sm1  = (const float*)d_in[11];
  const float* bm1  = (const float*)d_in[12];
  const float* Wm2  = (const float*)d_in[13];
  const float* sm2  = (const float*)d_in[14];
  const float* bm2  = (const float*)d_in[15];

  char* ws = (char*)d_ws;
  _Float16* whcv1 = (_Float16*)(ws + 0);          // 32*64*2   = 4096
  _Float16* whcv2 = (_Float16*)(ws + 4096);       // 32*64*2   = 4096
  _Float16* whm1  = (_Float16*)(ws + 8192);       // 16*32*2   = 1024
  _Float16* whm2  = (_Float16*)(ws + 9216);       // 32*160*2  = 10240
  _Float16* whcv3 = (_Float16*)(ws + 19456);      // 64*64*2   = 8192
  _Float16* xch   = (_Float16*)(ws + 32768);                         // 64x8192 f16
  _Float16* y1h   = (_Float16*)(ws + 32768 + 1048576);               // 32x8192
  _Float16* hhb   = (_Float16*)(ws + 32768 + 1048576 + 524288);      // 16x8192
  _Float16* hcol  = (_Float16*)(ws + 32768 + 1048576 + 524288 + 262144);          // 160x8192
  _Float16* ycat  = (_Float16*)(ws + 32768 + 1048576 + 524288 + 262144 + 2621440);// 64x8192

  // 1) pack weights into fragment-native fp16 A layout (K zero-padded)
  pack_weights_kernel<<<(32 * 64 + 255) / 256, 256, 0, stream>>>(Wcv1, whcv1, 32, 36, 64);
  pack_weights_kernel<<<(32 * 64 + 255) / 256, 256, 0, stream>>>(Wcv2, whcv2, 32, 36, 64);
  pack_weights_kernel<<<(16 * 32 + 255) / 256, 256, 0, stream>>>(Wm1,  whm1,  16, 32, 32);
  pack_weights_kernel<<<(32 * 160 + 255) / 256, 256, 0, stream>>>(Wm2, whm2,  32, 144, 160);
  pack_weights_kernel<<<(64 * 64 + 255) / 256, 256, 0, stream>>>(Wcv3, whcv3, 64, 64, 64);

  // 2) GLCM features + blocked fp16 concat activation matrix
  glcm_pack_kernel<<<1024, 256, 0, stream>>>(x, xch);

  // 3) cv1: [32x64]x[64x8192] -> y1h
  gemm_wmma_kernel<2, 2, false, true, false><<<64, 256, 0, stream>>>(
      whcv1, xch, scv1, bcv1, nullptr, y1h, 0, nullptr);
  // 4) cv2: -> ycat K-rows 32..63
  gemm_wmma_kernel<2, 2, false, true, false><<<64, 256, 0, stream>>>(
      whcv2, xch, scv2, bcv2, nullptr, ycat, 32, nullptr);
  // 5) bottleneck m1 (1x1): [16x32]x[32x8192] -> hhb
  gemm_wmma_kernel<1, 1, false, true, false><<<64, 256, 0, stream>>>(
      whm1, y1h, sm1, bm1, nullptr, hhb, 0, nullptr);
  // 6) im2col for the 3x3 conv
  im2col3x3_kernel<<<5120, 256, 0, stream>>>(hhb, hcol);
  // 7) bottleneck m2 (3x3) + shortcut: -> ycat K-rows 0..31
  gemm_wmma_kernel<2, 5, true, true, false><<<64, 256, 0, stream>>>(
      whm2, hcol, sm2, bm2, y1h, ycat, 0, nullptr);
  // 8) cv3: [64x64]x[64x8192] -> d_out (NCHW fp32)
  gemm_wmma_kernel<4, 2, false, false, true><<<64, 256, 0, stream>>>(
      whcv3, ycat, scv3, bcv3, nullptr, nullptr, 0, (float*)d_out);
}
